// LstmCellWithProjection_55327768708047
// MI455X (gfx1250) — matrix operand order repaired and checked by reading, not compiled, not run on hardware
//
#include <hip/hip_runtime.h>

typedef __bf16 bf16;
typedef __attribute__((ext_vector_type(16))) __bf16 v16bf;
typedef __attribute__((ext_vector_type(8)))  __bf16 v8bf;
typedef __attribute__((ext_vector_type(8)))  float  v8f;

#define B_   64
#define T_   256
#define I_   512
#define H_   512
#define C_   4096
#define G4_  (4 * C_)      /* 16384 gate rows        */
#define KZ_  (I_ + H_)     /* 1024 fused K dimension */
#define NWG  64
#define NTHR 512           /* 16 waves/WG -> 4 waves/SIMD for latency hiding */
#define NWAVE (NTHR / 32)

/* workspace layout (byte offsets, all 256-aligned) */
#define OFF_BAR ((size_t)0)
#define OFF_X   ((size_t)256)
#define OFF_WZ  (OFF_X   + (size_t)T_ * B_ * I_ * 2)   /* x in [T][B][I] bf16 */
#define OFF_WP  (OFF_WZ  + (size_t)G4_ * KZ_ * 2)      /* [16384][1024] bf16  */
#define OFF_HBF (OFF_WP  + (size_t)H_ * C_ * 2)        /* [512][4096] bf16    */
#define OFF_HF  (OFF_HBF + (size_t)B_ * H_ * 2)        /* h bf16              */
#define OFF_CF  (OFF_HF  + (size_t)B_ * H_ * 4)        /* h f32               */
#define OFF_G   (OFF_CF  + (size_t)B_ * C_ * 4)        /* c f32               */
#define WS_END  (OFF_G   + (size_t)B_ * C_ * 2)        /* gated o*tanh(c) bf16*/

#define YSZ ((size_t)B_ * T_ * H_)

union V16 { v16bf v; v8bf h[2]; };

/* A fragment (16x32 bf16): lane 0-15 = row m0+lane, K {0..7,16..23};
   lane 16-31 = row m0+lane-16, K {8..15,24..31}  (ISA 7.12.2)          */
__device__ __forceinline__ v16bf load_frag_a(const bf16* __restrict__ base,
                                             int ld, int row0, int k0, int lane) {
  const int hi = lane >> 4;
  const bf16* p = base + (size_t)(row0 + (lane & 15)) * ld + k0 + hi * 8;
  V16 u;
  u.h[0] = *(const v8bf*)(p);
  u.h[1] = *(const v8bf*)(p + 16);
  return u.v;
}

/* B fragment (32x16 bf16), B(k,n) = W[n0+n][k] with W row-major over K:
   lane 0-15 = col n, K {0..7,16..23}; lane 16-31 = col n-16, K {8..15,24..31}
   -> two 16-byte pieces per lane. Optionally prefetch next k-step.     */
template <bool PF>
__device__ __forceinline__ v16bf load_frag_b(const bf16* __restrict__ W,
                                             int ld, int n0, int k0, int lane) {
  const int hi = lane >> 4;
  const bf16* p = W + (size_t)(n0 + (lane & 15)) * ld + k0 + hi * 8;
  V16 u;
  u.h[0] = *(const v8bf*)(p);
  u.h[1] = *(const v8bf*)(p + 16);
  if (PF) __builtin_prefetch(p + 32, 0, 3);   /* next k-step -> global_prefetch_b8 */
  return u.v;
}

__device__ __forceinline__ float sigm(float x)  { return 1.f / (1.f + __expf(-x)); }
__device__ __forceinline__ float clip3(float x) { return fminf(fmaxf(x, -3.f), 3.f); }

/* monotonic-counter grid barrier (persistent kernel, 64 co-resident WGs) */
__device__ __forceinline__ void grid_barrier(unsigned* bar, unsigned target) {
  __threadfence();                       /* push my stores to device scope */
  __syncthreads();
  if (threadIdx.x == 0) {
    __hip_atomic_fetch_add(bar, 1u, __ATOMIC_ACQ_REL, __HIP_MEMORY_SCOPE_AGENT);
    while (__hip_atomic_load(bar, __ATOMIC_ACQUIRE, __HIP_MEMORY_SCOPE_AGENT) < target)
      __builtin_amdgcn_s_sleep(2);
  }
  __syncthreads();
  __threadfence();                       /* drop stale lines before reading peers */
}

/* ---------------- phase 0: bf16 conversion / init ---------------- */
__global__ __launch_bounds__(256) void prep_kernel(
    const float* __restrict__ inputs, const float* __restrict__ W_in,
    const float* __restrict__ W_state, const float* __restrict__ W_proj,
    bf16* __restrict__ Xbf, bf16* __restrict__ Wz, bf16* __restrict__ Wp,
    bf16* __restrict__ Hbf, float* __restrict__ Hf32, float* __restrict__ Cf32,
    unsigned* __restrict__ bar) {
  const int stride = gridDim.x * blockDim.x;
  const int g0 = blockIdx.x * blockDim.x + threadIdx.x;
  /* fused weight matrix Wz[g][k]: k<512 -> W_in, k>=512 -> W_state */
  for (int idx = g0; idx < G4_ * KZ_; idx += stride) {
    const int row = idx >> 10, k = idx & (KZ_ - 1);
    const float v = (k < I_) ? W_in[(size_t)row * I_ + k]
                             : W_state[(size_t)row * H_ + (k - I_)];
    Wz[idx] = (bf16)v;
  }
  /* inputs [B][T][I] f32 -> Xbf [T][B][I] bf16 */
  for (int idx = g0; idx < T_ * B_ * I_; idx += stride) {
    const int i = idx & (I_ - 1);
    const int b = (idx >> 9) & (B_ - 1);
    const int t = idx >> 15;
    Xbf[idx] = (bf16)inputs[((size_t)b * T_ + t) * I_ + i];
  }
  for (int idx = g0; idx < H_ * C_; idx += stride) Wp[idx] = (bf16)W_proj[idx];
  for (int idx = g0; idx < B_ * H_; idx += stride) { Hbf[idx] = (bf16)0.f; Hf32[idx] = 0.f; }
  for (int idx = g0; idx < B_ * C_; idx += stride) Cf32[idx] = 0.f;
  if (g0 == 0) *bar = 0u;
}

/* ---------------- persistent fused LSTM scan ---------------- */
__global__ __launch_bounds__(NTHR, 1) void lstm_persistent(
    const float* __restrict__ bstate, const int* __restrict__ lens,
    const bf16* __restrict__ Xbf, const bf16* __restrict__ Wz,
    const bf16* __restrict__ Wp, bf16* __restrict__ Hbf,
    float* __restrict__ Hf32, float* __restrict__ Cf32,
    bf16* __restrict__ G, float* __restrict__ out, unsigned* __restrict__ bar) {
  const int lane = threadIdx.x & 31;
  const int wave = threadIdx.x >> 5;
  const int wg   = blockIdx.x;
  const int gw   = wg * NWAVE + wave;

  /* z-phase: each wave owns one (m,cn) tile and all 4 gates of it */
  const int m0   = (wave & 3) * 16;
  const int cn   = wg * 4 + (wave >> 2);      /* 64 WGs x 4 = 256 cn tiles */
  const int hi8  = (lane >> 4) * 8;
  const int ncol = lane & 15;
  const int col  = cn * 16 + ncol;

  /* ---- t-invariant values hoisted out of the scan ---- */
  float bias[4];
#pragma unroll
  for (int g = 0; g < 4; ++g) bias[g] = bstate[g * C_ + col];
  int lenr[8];
#pragma unroll
  for (int r = 0; r < 8; ++r) lenr[r] = lens[m0 + r + hi8];

  /* projection tile (first 128 waves) + its t-invariants */
  const bool do_proj = gw < (B_ / 16) * (H_ / 16);
  const int pm0 = (gw & 3) * 16;
  const int pn0 = (gw >> 2) * 16;
  int plen[8];
#pragma unroll
  for (int r = 0; r < 8; ++r) plen[r] = lens[(pm0 + r + hi8) & (B_ - 1)];

  unsigned epoch = 0;

  for (int t = 0; t < T_; ++t) {
    const bf16* Xt = Xbf + (size_t)t * (B_ * I_);

    /* ---- gate GEMM: z[64,16384] = [x_t ; h] (64x1024) x Wz^T ----
       batch 4 B loads, then 4 WMMAs: partial loadcnt waits overlap
       the tail loads with matrix math.                              */
    v8f acc[4];
#pragma unroll
    for (int g = 0; g < 4; ++g) acc[g] = {};

#pragma unroll 1
    for (int k0 = 0; k0 < I_; k0 += 32) {            /* x_t part, K 0..511 */
      const v16bf af = load_frag_a(Xt, I_, m0, k0, lane);
      v16bf bq[4];
#pragma unroll
      for (int g = 0; g < 4; ++g)
        bq[g] = load_frag_b<true>(Wz, KZ_, g * C_ + cn * 16, k0, lane);
#pragma unroll
      for (int g = 0; g < 4; ++g)
        acc[g] = __builtin_amdgcn_wmma_f32_16x16x32_bf16(
            false, af, false, bq[g], (short)0, acc[g], false, false);
      __builtin_amdgcn_sched_barrier(0);
    }
#pragma unroll 1
    for (int k0 = 0; k0 < H_; k0 += 32) {            /* h part, K 512..1023 */
      const v16bf af = load_frag_a(Hbf, H_, m0, k0, lane);
      v16bf bq[4];
#pragma unroll
      for (int g = 0; g < 4; ++g)
        bq[g] = load_frag_b<true>(Wz, KZ_, g * C_ + cn * 16, I_ + k0, lane);
#pragma unroll
      for (int g = 0; g < 4; ++g)
        acc[g] = __builtin_amdgcn_wmma_f32_16x16x32_bf16(
            false, af, false, bq[g], (short)0, acc[g], false, false);
      __builtin_amdgcn_sched_barrier(0);
    }

    /* ---- fused gates + cell update + o*tanh(c) (registers only) ---- */
#pragma unroll
    for (int r = 0; r < 8; ++r) {
      const int b = m0 + r + hi8;                 /* D layout: M = r + 8*(lane>=16) */
      const float ig = sigm(acc[0][r] + bias[0]);
      const float fg = sigm(acc[1][r] + bias[1]);
      const float gg = tanhf(acc[2][r] + bias[2]);
      const float og = sigm(acc[3][r] + bias[3]);
      const size_t ci = (size_t)b * C_ + col;
      const float cold = Cf32[ci];
      const float cnew = clip3(ig * gg + fg * cold);
      if (lenr[r] > t) Cf32[ci] = cnew;           /* c is thread-private across steps */
      G[ci] = (bf16)(og * tanhf(cnew));
    }
    epoch += 1;
    grid_barrier(bar, epoch * NWG);

    /* ---- projection: out[64,512] = G (64x4096) x Wp^T ----
       2-way K unroll, two accumulators -> no serial WMMA chain */
    if (do_proj) {                                  /* 128 tiles, 1 per wave */
      v8f p0 = {}, p1 = {};
#pragma unroll 1
      for (int k0 = 0; k0 < C_; k0 += 64) {
        const v16bf a0 = load_frag_a(G, C_, pm0, k0, lane);
        const v16bf b0 = load_frag_b<false>(Wp, C_, pn0, k0, lane);
        const v16bf a1 = load_frag_a(G, C_, pm0, k0 + 32, lane);
        const v16bf b1 = load_frag_b<false>(Wp, C_, pn0, k0 + 32, lane);
        p0 = __builtin_amdgcn_wmma_f32_16x16x32_bf16(
            false, a0, false, b0, (short)0, p0, false, false);
        p1 = __builtin_amdgcn_wmma_f32_16x16x32_bf16(
            false, a1, false, b1, (short)0, p1, false, false);
        __builtin_amdgcn_sched_barrier(0);
      }
      const v8f p = p0 + p1;
      const int n = pn0 + ncol;
#pragma unroll
      for (int r = 0; r < 8; ++r) {
        const int b = pm0 + r + hi8;
        const float val  = clip3(p[r]);
        const bool  mk   = plen[r] > t;
        const size_t hidx = (size_t)b * H_ + n;
        const float hold = Hf32[hidx];
        const float hnew = mk ? val : hold;
        Hf32[hidx] = hnew;
        Hbf[hidx]  = (bf16)hnew;
        out[((size_t)b * T_ + t) * H_ + n] = mk ? val : 0.f;
      }
    }
    epoch += 1;
    grid_barrier(bar, epoch * NWG);
  }

  /* ---- final states: (1,B,H) then (1,B,C) appended after y ---- */
  for (int idx = blockIdx.x * NTHR + threadIdx.x; idx < B_ * H_; idx += NWG * NTHR)
    out[YSZ + idx] = Hf32[idx];
  for (int idx = blockIdx.x * NTHR + threadIdx.x; idx < B_ * C_; idx += NWG * NTHR)
    out[YSZ + B_ * H_ + idx] = Cf32[idx];
}

extern "C" void kernel_launch(void* const* d_in, const int* in_sizes, int n_in,
                              void* d_out, int out_size, void* d_ws, size_t ws_size,
                              hipStream_t stream) {
  (void)in_sizes; (void)n_in; (void)out_size; (void)ws_size;
  const float* inputs  = (const float*)d_in[0];
  const int*   lens    = (const int*)  d_in[1];
  const float* W_in    = (const float*)d_in[2];
  const float* W_state = (const float*)d_in[3];
  const float* b_state = (const float*)d_in[4];
  const float* W_proj  = (const float*)d_in[5];

  char* ws = (char*)d_ws;
  unsigned* bar = (unsigned*)(ws + OFF_BAR);
  bf16*  Xbf  = (bf16*) (ws + OFF_X);
  bf16*  Wz   = (bf16*) (ws + OFF_WZ);
  bf16*  Wp   = (bf16*) (ws + OFF_WP);
  bf16*  Hbf  = (bf16*) (ws + OFF_HBF);
  float* Hf32 = (float*)(ws + OFF_HF);
  float* Cf32 = (float*)(ws + OFF_CF);
  bf16*  G    = (bf16*) (ws + OFF_G);

  prep_kernel<<<2048, 256, 0, stream>>>(inputs, W_in, W_state, W_proj,
                                        Xbf, Wz, Wp, Hbf, Hf32, Cf32, bar);
  lstm_persistent<<<NWG, NTHR, 0, stream>>>(b_state, lens, Xbf, Wz, Wp, Hbf,
                                            Hf32, Cf32, G, (float*)d_out, bar);
}
